// RPN_66898410603005
// MI455X (gfx1250) — compile-verified, matrix operand order checked by reading, not run on hardware
//
#include <hip/hip_runtime.h>
#include <hip/hip_bf16.h>
#include <math.h>

// ---------------------------------------------------------------------------
// Types
// ---------------------------------------------------------------------------
typedef __attribute__((ext_vector_type(16))) _Float16 v16h;
typedef __attribute__((ext_vector_type(8)))  float    v8f;
typedef __attribute__((ext_vector_type(4)))  float    f32x4;
typedef __attribute__((ext_vector_type(4)))  unsigned int u32x4;
typedef __attribute__((ext_vector_type(8)))  int      i32x8;
typedef __attribute__((ext_vector_type(4)))  int      i32x4;

// ---------------------------------------------------------------------------
// Problem constants (match the reference)
// ---------------------------------------------------------------------------
#define C_IN    128
#define KDIM    1152          // 128 * 3 * 3
#define KPAD    1160          // padded K stride in LDS (2320 B/row, 16B aligned)
#define A_NUM   3
#define N_IMAGES 2
#define R_TOT   130944        // total anchors per image over 5 levels
#define POST_K  1000
#define NMS_TH  0.7f
#define IMG_WF  1024.0f
#define IMG_HF  512.0f
#define SCALE_CLAMP_F 4.135166556742356f   // log(1000/16)

#define TDM_CHUNK 64          // channels per TENSOR_LOAD_TO_LDS chunk
#define TILE_W    18          // 16 px + 3x3 halo
#define TILE_H    3

__device__ const int   d_lvl_off[6] = {0, 98304, 122880, 129024, 130560, 130944};
__device__ const int   d_lvl_w[5]   = {256, 128, 64, 32, 16};
__device__ const int   d_lvl_s[5]   = {4, 8, 16, 32, 64};
__device__ const float d_lvl_sz[5]  = {32.f, 64.f, 128.f, 256.f, 512.f};

// ---------------------------------------------------------------------------
// Weight repack (f32 -> f16, GEMM-friendly layouts)
// ---------------------------------------------------------------------------
__global__ void convert_convw_kernel(const float* __restrict__ cw,
                                     _Float16* __restrict__ wA) {
  int idx = blockIdx.x * 256 + threadIdx.x;
  if (idx >= C_IN * KDIM) return;
  int oc  = idx / KDIM;
  int k   = idx - oc * KDIM;
  int cin = k / 9;
  int r9  = k - cin * 9;
  // conv_w is [O=128][I=128][3][3]; K index = cin*9 + (dy+1)*3 + (dx+1)
  wA[idx] = (_Float16)cw[((size_t)oc * C_IN + cin) * 9 + r9];
}

__global__ void convert_headw_kernel(const float* __restrict__ ow,
                                     const float* __restrict__ dw,
                                     _Float16* __restrict__ wH) {
  int idx = blockIdx.x * 256 + threadIdx.x;
  if (idx >= 16 * C_IN) return;
  int hc = idx / C_IN;
  int k  = idx - hc * C_IN;
  float v = 0.0f;
  if (hc < 3)       v = ow[hc * C_IN + k];          // objectness rows 0..2
  else if (hc < 15) v = dw[(hc - 3) * C_IN + k];    // delta rows 3..14 (a*4+c)
  wH[idx] = (_Float16)v;                            // row 15 = zero pad
}

__global__ void zero_losses_kernel(float* __restrict__ losses) {
  losses[0] = 0.0f;
  losses[1] = 0.0f;
}

// ---------------------------------------------------------------------------
// Fused RPN head: TDM-staged im2col + 3x3 conv (WMMA f16) + ReLU + 1x1 heads
// Block = 256 threads = 8 waves. Block tile = 16 pixels x 128 out channels.
// Input tile (3 rows x 18 cols x 128 cin, f32) is DMA'd to LDS by the
// Tensor Data Mover in two 64-channel chunks, then repacked f32->f16 as
// im2col entirely inside LDS. Conv = 36 WMMAs/wave; heads = 4 WMMAs (wave 0).
// ---------------------------------------------------------------------------
__global__ __launch_bounds__(256)
void rpn_head_wmma_kernel(const float* __restrict__ feat,      // [N,128,H,W]
                          const _Float16* __restrict__ wA,     // [128][1152]
                          const float* __restrict__ conv_b,    // [128]
                          const _Float16* __restrict__ wHead,  // [16][128]
                          const float* __restrict__ obj_b,     // [3]
                          const float* __restrict__ delta_b,   // [12]
                          float* __restrict__ logits_out,      // [N,R_TOT]
                          float* __restrict__ deltas_out,      // [N,R_TOT,4]
                          int H, int W, int lvl_off) {
  __shared__ __align__(16) float    sStage[TDM_CHUNK * TILE_H * TILE_W]; // 13824 B
  __shared__ __align__(16) _Float16 sB[16 * KPAD];                       // 37120 B
  __shared__ __align__(16) _Float16 sH[16 * 136];                        //  4352 B

  const int n        = blockIdx.y;
  const int tileBase = blockIdx.x * 16;
  const int HW       = H * W;
  const int tid      = threadIdx.x;

  const int hrow = tileBase / W;
  const int wb   = tileBase - hrow * W;
  const int h0   = (hrow > 0) ? hrow - 1 : 0;   // staged-tile origin (clamped)
  const int w0   = (wb > 0) ? wb - 1 : 0;

  // ---- Stage + repack the input patch in two TDM chunks of 64 channels ----
  for (int c0 = 0; c0 < C_IN; c0 += TDM_CHUNK) {
#if __has_builtin(__builtin_amdgcn_tensor_load_to_lds)
    if (tid < 32) {
      // Tensor DMA descriptor (D#): 3D tile [z=64 cin][y=3 rows][x=18 cols],
      // element = 4B, strides (HW, W, 1) in elements.
      uint64_t gaddr = (uint64_t)(uintptr_t)(feat +
          ((size_t)n * C_IN + c0) * HW + (size_t)h0 * W + w0);
      unsigned ldsa = (unsigned)(uintptr_t)(void*)sStage;

      u32x4 g0;
      g0[0] = 1u;                                   // count=1 (valid), user mode
      g0[1] = ldsa;                                 // lds_addr (bytes)
      g0[2] = (unsigned)(gaddr & 0xffffffffu);      // global_addr[31:0]
      g0[3] = (unsigned)((gaddr >> 32) & 0x01ffffffu) | 0x80000000u; // [56:32]|type=2

      unsigned td0 = (unsigned)W;                   // tensor_dim0
      unsigned td1 = (unsigned)H;                   // tensor_dim1
      unsigned s0  = (unsigned)W;                   // tensor_dim0_stride
      uint64_t s1  = (uint64_t)(unsigned)HW;        // tensor_dim1_stride
      i32x8 g1;
      g1[0] = (int)(2u << 16);                      // wg_mask=0, data_size=2 (4B)
      g1[1] = (int)((td0 & 0xffffu) << 16);         // barrier_addr=0 | td0[15:0]
      g1[2] = (int)((td0 >> 16) | ((td1 & 0xffffu) << 16));
      g1[3] = (int)((td1 >> 16) | ((unsigned)TILE_W << 16));      // tile_dim0=18
      g1[4] = (int)((unsigned)TILE_H | ((unsigned)TDM_CHUNK << 16)); // tdim1=3,tdim2=64
      g1[5] = (int)s0;                              // stride0[31:0]
      g1[6] = (int)((unsigned)(s1 & 0xffffu) << 16);// stride0[47:32]=0 | stride1[15:0]
      g1[7] = (int)(unsigned)(s1 >> 16);            // stride1[47:16]

      i32x4 g2;
      g2[0] = TDM_CHUNK;                            // tensor_dim2
      g2[1] = 0; g2[2] = 0; g2[3] = 0;
      i32x4 g3; g3[0] = 0; g3[1] = 0; g3[2] = 0; g3[3] = 0;
      i32x8 g4;                                     // extended group (unused)
      g4[0] = 0; g4[1] = 0; g4[2] = 0; g4[3] = 0;
      g4[4] = 0; g4[5] = 0; g4[6] = 0; g4[7] = 0;

      __builtin_amdgcn_tensor_load_to_lds(g0, g1, g2, g3, g4, 0);
      __builtin_amdgcn_s_wait_tensorcnt(0);
    }
    __syncthreads();

    // repack staged f32 -> f16 im2col rows for channels [c0, c0+64)
    for (int idx = tid; idx < 16 * TDM_CHUNK * 9; idx += 256) {
      int px  = idx / (TDM_CHUNK * 9);
      int rem = idx - px * (TDM_CHUNK * 9);
      int cl  = rem / 9;                    // local channel
      int r9  = rem - cl * 9;
      int dy  = r9 / 3 - 1;
      int dx  = r9 - (r9 / 3) * 3 - 1;
      int pid = tileBase + px;
      int hh  = hrow + dy;
      int ww  = wb + px + dx;
      int sy  = hh - h0;
      int sx  = ww - w0;
      float v = 0.0f;
      if (pid < HW && hh >= 0 && hh < H && ww >= 0 && ww < W &&
          sy >= 0 && sy < TILE_H && sx >= 0 && sx < TILE_W)
        v = sStage[(cl * TILE_H + sy) * TILE_W + sx];
      sB[px * KPAD + (c0 + cl) * 9 + r9] = (_Float16)v;
    }
    __syncthreads();
#else
    // Fallback: direct global gather (no TDM builtin on this toolchain)
    for (int idx = tid; idx < 16 * TDM_CHUNK * 9; idx += 256) {
      int px  = idx / (TDM_CHUNK * 9);
      int rem = idx - px * (TDM_CHUNK * 9);
      int cl  = rem / 9;
      int r9  = rem - cl * 9;
      int dy  = r9 / 3 - 1;
      int dx  = r9 - (r9 / 3) * 3 - 1;
      int pid = tileBase + px;
      int h   = pid / W;
      int w   = pid - h * W;
      int hh  = h + dy, ww = w + dx;
      float v = 0.0f;
      if (pid < HW && hh >= 0 && hh < H && ww >= 0 && ww < W)
        v = feat[((size_t)n * C_IN + c0 + cl) * HW + (size_t)hh * W + ww];
      sB[px * KPAD + (c0 + cl) * 9 + r9] = (_Float16)v;
    }
    __syncthreads();
#endif
  }

  const int wave  = tid >> 5;
  const int lane  = tid & 31;
  const int mrow  = wave * 16 + (lane & 15);  // out-channel row of A fragment
  const int khalf = (lane >> 4) * 8;          // per-lane K sub-offset (A/B layout)
  const int ncol  = lane & 15;                // pixel column of B fragment

  // ---- Main conv GEMM: 36 x v_wmma_f32_16x16x32_f16 per wave ----
  v8f acc = {};
  const _Float16* arow = wA + (size_t)mrow * KDIM + khalf;
  const _Float16* brow = sB + ncol * KPAD + khalf;
  for (int k0 = 0; k0 < KDIM; k0 += 32) {
    __builtin_prefetch(arow + k0 + 128, 0, 1);
    union { v16h v; f32x4 q[2]; } ua, ub;
    ua.q[0] = *(const f32x4*)(arow + k0);
    ua.q[1] = *(const f32x4*)(arow + k0 + 16);
    ub.q[0] = *(const f32x4*)(brow + k0);
    ub.q[1] = *(const f32x4*)(brow + k0 + 16);
    acc = __builtin_amdgcn_wmma_f32_16x16x32_f16(false, ua.v, false, ub.v,
                                                 (short)0, acc, false, false);
  }

  // ---- bias + ReLU, park hidden activations in LDS as f16 [px][oc] ----
  {
    int px    = lane & 15;
    int mbase = (lane >> 4) * 8;
    for (int r = 0; r < 8; ++r) {
      int oc   = wave * 16 + mbase + r;
      float hv = acc[r] + conv_b[oc];
      sH[px * 136 + oc] = (_Float16)(hv > 0.0f ? hv : 0.0f);
    }
  }
  __syncthreads();

  // ---- 1x1 heads: rows 0..2 obj, 3..14 deltas, K=128 -> 4 WMMAs (wave 0) ----
  if (wave == 0) {
    v8f hacc = {};
    const _Float16* ap = wHead + (lane & 15) * C_IN + khalf;
    const _Float16* bp = sH + ncol * 136 + khalf;
    for (int k0 = 0; k0 < C_IN; k0 += 32) {
      union { v16h v; f32x4 q[2]; } ua, ub;
      ua.q[0] = *(const f32x4*)(ap + k0);
      ua.q[1] = *(const f32x4*)(ap + k0 + 16);
      ub.q[0] = *(const f32x4*)(bp + k0);
      ub.q[1] = *(const f32x4*)(bp + k0 + 16);
      hacc = __builtin_amdgcn_wmma_f32_16x16x32_f16(false, ua.v, false, ub.v,
                                                    (short)0, hacc, false, false);
    }
    int px  = lane & 15;
    int pid = tileBase + px;
    if (pid < HW) {
      int mbase = (lane >> 4) * 8;
      size_t base = (size_t)n * R_TOT + lvl_off + (size_t)pid * A_NUM;
      for (int r = 0; r < 8; ++r) {
        int hc  = mbase + r;
        float v = hacc[r];
        if (hc < 3) {
          logits_out[base + hc] = v + obj_b[hc];           // (h,w,a) order
        } else if (hc < 15) {
          int d = hc - 3;                                  // d = a*4 + c
          deltas_out[(base + (d >> 2)) * 4 + (d & 3)] = v + delta_b[d];
        }
      }
    }
  }
}

// ---------------------------------------------------------------------------
// Anchor generation helper (level decomposition on the fly)
// ---------------------------------------------------------------------------
__device__ __forceinline__ void anchor_of(int r, float* b) {
  int l = 0;
  if (r >= d_lvl_off[1]) l = 1;
  if (r >= d_lvl_off[2]) l = 2;
  if (r >= d_lvl_off[3]) l = 3;
  if (r >= d_lvl_off[4]) l = 4;
  int rr  = r - d_lvl_off[l];
  int a   = rr % 3;
  int pid = rr / 3;
  int Wl  = d_lvl_w[l];
  int h   = pid / Wl;
  int w   = pid - h * Wl;
  float ratio = (a == 0) ? 0.5f : (a == 1 ? 1.0f : 2.0f);
  float area  = d_lvl_sz[l] * d_lvl_sz[l];
  float wa    = sqrtf(area / ratio);
  float ha    = wa * ratio;
  float cx    = (float)(w * d_lvl_s[l]);
  float cy    = (float)(h * d_lvl_s[l]);
  b[0] = cx - 0.5f * wa; b[1] = cy - 0.5f * ha;
  b[2] = cx + 0.5f * wa; b[3] = cy + 0.5f * ha;
}

// ---------------------------------------------------------------------------
// apply_deltas + clip + validity -> proposal boxes & scores
// ---------------------------------------------------------------------------
__global__ __launch_bounds__(256)
void proposals_kernel(const float* __restrict__ logits,
                      const float* __restrict__ deltas,
                      float* __restrict__ boxes,
                      float* __restrict__ scores) {
  int idx = blockIdx.x * 256 + threadIdx.x;
  if (idx >= N_IMAGES * R_TOT) return;
  int r = idx % R_TOT;
  float anc[4];
  anchor_of(r, anc);
  float w  = anc[2] - anc[0], h = anc[3] - anc[1];
  float cx = anc[0] + 0.5f * w, cy = anc[1] + 0.5f * h;
  const float* d = deltas + (size_t)idx * 4;
  float dw = fminf(d[2], SCALE_CLAMP_F);
  float dh = fminf(d[3], SCALE_CLAMP_F);
  float pcx = d[0] * w + cx, pcy = d[1] * h + cy;
  float pw = expf(dw) * w, ph = expf(dh) * h;
  float x0 = fminf(fmaxf(pcx - 0.5f * pw, 0.0f), IMG_WF);
  float y0 = fminf(fmaxf(pcy - 0.5f * ph, 0.0f), IMG_HF);
  float x1 = fminf(fmaxf(pcx + 0.5f * pw, 0.0f), IMG_WF);
  float y1 = fminf(fmaxf(pcy + 0.5f * ph, 0.0f), IMG_HF);
  float* bo = boxes + (size_t)idx * 4;
  bo[0] = x0; bo[1] = y0; bo[2] = x1; bo[3] = y1;
  bool valid = ((x1 - x0) > 0.0f) && ((y1 - y0) > 0.0f);
  scores[idx] = valid ? logits[idx] : -INFINITY;
}

// ---------------------------------------------------------------------------
// top-k with greedy NMS against the kept set (one block per image)
// ---------------------------------------------------------------------------
__global__ __launch_bounds__(256)
void topk_nms_kernel(float* __restrict__ scores,      // [N,R] (consumed)
                     const float* __restrict__ boxes, // [N,R,4]
                     float* __restrict__ out_boxes,   // [N,1000,4]
                     float* __restrict__ out_scores)  // [N,1000]
{
  const int n = blockIdx.x;
  float* sc = scores + (size_t)n * R_TOT;
  const float* bx = boxes + (size_t)n * R_TOT * 4;

  __shared__ float smax[256];
  __shared__ int   sidx[256];
  __shared__ float kept[POST_K * 4];
  __shared__ int   nkept;
  if (threadIdx.x == 0) nkept = 0;

  for (int k = 0; k < POST_K; ++k) {
    float best = -INFINITY; int bi = -1;
    for (int r = threadIdx.x; r < R_TOT; r += 256) {
      float s = sc[r];
      if (s > best) { best = s; bi = r; }
    }
    smax[threadIdx.x] = best;
    sidx[threadIdx.x] = bi;
    __syncthreads();
    for (int off = 128; off > 0; off >>= 1) {
      if (threadIdx.x < off && smax[threadIdx.x + off] > smax[threadIdx.x]) {
        smax[threadIdx.x] = smax[threadIdx.x + off];
        sidx[threadIdx.x] = sidx[threadIdx.x + off];
      }
      __syncthreads();
    }
    if (threadIdx.x == 0) {
      int r = sidx[0];
      float s = smax[0];
      float b0 = 0.f, b1 = 0.f, b2 = 0.f, b3 = 0.f;
      if (r >= 0) { b0 = bx[r*4]; b1 = bx[r*4+1]; b2 = bx[r*4+2]; b3 = bx[r*4+3]; }
      bool sup = false;
      float area = (b2 - b0) * (b3 - b1);
      for (int j = 0; j < nkept; ++j) {
        float kx0 = kept[j*4], ky0 = kept[j*4+1], kx1 = kept[j*4+2], ky1 = kept[j*4+3];
        float ix0 = fmaxf(b0, kx0), iy0 = fmaxf(b1, ky0);
        float ix1 = fminf(b2, kx1), iy1 = fminf(b3, ky1);
        float iw = fmaxf(ix1 - ix0, 0.f), ih = fmaxf(iy1 - iy0, 0.f);
        float inter = iw * ih;
        float ka = (kx1 - kx0) * (ky1 - ky0);
        float u = area + ka - inter;
        float iou = (inter > 0.f && u > 0.f) ? inter / u : 0.f;
        if (iou > NMS_TH) { sup = true; break; }
      }
      size_t o = (size_t)n * POST_K + k;
      out_boxes[o*4+0] = b0; out_boxes[o*4+1] = b1;
      out_boxes[o*4+2] = b2; out_boxes[o*4+3] = b3;
      out_scores[o] = (sup || r < 0) ? -INFINITY : s;
      if (!sup && r >= 0 && nkept < POST_K) {
        kept[nkept*4+0] = b0; kept[nkept*4+1] = b1;
        kept[nkept*4+2] = b2; kept[nkept*4+3] = b3;
        nkept++;
      }
      if (r >= 0) sc[r] = -INFINITY;
    }
    __syncthreads();
  }
}

// ---------------------------------------------------------------------------
// IoU matching + BCE / L1 loss reduction (atomics into d_out[10000..10001])
// ---------------------------------------------------------------------------
__global__ __launch_bounds__(256)
void loss_kernel(const float* __restrict__ gt,       // [N,32,4]
                 const float* __restrict__ logits,   // [N,R]
                 const float* __restrict__ deltas,   // [N,R,4]
                 float* __restrict__ out_losses)     // [2]
{
  __shared__ float sgt[32 * 4];
  const int n = blockIdx.y;
  if (threadIdx.x < 128) sgt[threadIdx.x] = gt[n * 128 + threadIdx.x];
  __syncthreads();

  int r = blockIdx.x * 256 + threadIdx.x;
  float cls = 0.f, loc = 0.f;
  if (r < R_TOT) {
    float anc[4];
    anchor_of(r, anc);
    float aarea = (anc[2] - anc[0]) * (anc[3] - anc[1]);
    float best = 0.f; int bi = 0;
    for (int g = 0; g < 32; ++g) {
      float gx0 = sgt[g*4], gy0 = sgt[g*4+1], gx1 = sgt[g*4+2], gy1 = sgt[g*4+3];
      float ix0 = fmaxf(anc[0], gx0), iy0 = fmaxf(anc[1], gy0);
      float ix1 = fminf(anc[2], gx1), iy1 = fminf(anc[3], gy1);
      float iw = fmaxf(ix1 - ix0, 0.f), ih = fmaxf(iy1 - iy0, 0.f);
      float inter = iw * ih;
      float ga = (gx1 - gx0) * (gy1 - gy0);
      float u = aarea + ga - inter;
      float iou = inter > 0.f ? inter / u : 0.f;
      if (iou > best) { best = iou; bi = g; }
    }
    int label = (best >= 0.7f) ? 1 : ((best >= 0.3f) ? -1 : 0);
    size_t o = (size_t)n * R_TOT + r;
    float x  = logits[o];
    float fg = (label == 1) ? 1.f : 0.f;
    if (label >= 0)
      cls = fmaxf(x, 0.f) - x * fg + log1pf(expf(-fabsf(x)));
    if (label == 1) {
      float gx0 = sgt[bi*4], gy0 = sgt[bi*4+1], gx1 = sgt[bi*4+2], gy1 = sgt[bi*4+3];
      float sw = anc[2] - anc[0], sh = anc[3] - anc[1];
      float scx = anc[0] + 0.5f * sw, scy = anc[1] + 0.5f * sh;
      float tw = gx1 - gx0, th = gy1 - gy0;
      float tcx = gx0 + 0.5f * tw, tcy = gy0 + 0.5f * th;
      float gd0 = (tcx - scx) / sw, gd1 = (tcy - scy) / sh;
      float gd2 = logf(tw / sw),    gd3 = logf(th / sh);
      const float* dd = deltas + o * 4;
      loc = fabsf(dd[0]-gd0) + fabsf(dd[1]-gd1) + fabsf(dd[2]-gd2) + fabsf(dd[3]-gd3);
    }
  }

  __shared__ float scl[256], slo[256];
  scl[threadIdx.x] = cls; slo[threadIdx.x] = loc;
  __syncthreads();
  for (int off = 128; off > 0; off >>= 1) {
    if (threadIdx.x < off) {
      scl[threadIdx.x] += scl[threadIdx.x + off];
      slo[threadIdx.x] += slo[threadIdx.x + off];
    }
    __syncthreads();
  }
  if (threadIdx.x == 0) {
    const float inv = 1.0f / (256.0f * (float)N_IMAGES);
    atomicAdd(&out_losses[0], scl[0] * inv);
    atomicAdd(&out_losses[1], slo[0] * inv);
  }
}

// ---------------------------------------------------------------------------
// Launch
// ---------------------------------------------------------------------------
extern "C" void kernel_launch(void* const* d_in, const int* in_sizes, int n_in,
                              void* d_out, int out_size, void* d_ws, size_t ws_size,
                              hipStream_t stream) {
  const float* feats[5] = {(const float*)d_in[0], (const float*)d_in[1],
                           (const float*)d_in[2], (const float*)d_in[3],
                           (const float*)d_in[4]};
  const float* gt      = (const float*)d_in[5];
  const float* conv_w  = (const float*)d_in[6];
  const float* conv_b  = (const float*)d_in[7];
  const float* obj_w   = (const float*)d_in[8];
  const float* obj_b   = (const float*)d_in[9];
  const float* delta_w = (const float*)d_in[10];
  const float* delta_b = (const float*)d_in[11];

  // Workspace layout (all 16B aligned offsets)
  char* ws = (char*)d_ws;
  _Float16* wA = (_Float16*)ws;                                   // 294912 B
  _Float16* wH = (_Float16*)(ws + 294912);                        //   4096 B
  float* logits = (float*)(ws + 299008);                          // 1047552 B
  float* deltas = (float*)(ws + 299008 + 1047552);                // 4190208 B
  float* boxes  = (float*)(ws + 299008 + 1047552 + 4190208);      // 4190208 B
  float* scores = (float*)(ws + 299008 + 1047552 + 2*4190208);    // 1047552 B

  float* out_boxes  = (float*)d_out;                 // [2,1000,4]
  float* out_scores = (float*)d_out + N_IMAGES * POST_K * 4;
  float* out_losses = (float*)d_out + N_IMAGES * POST_K * 5;

  convert_convw_kernel<<<(C_IN * KDIM + 255) / 256, 256, 0, stream>>>(conv_w, wA);
  convert_headw_kernel<<<(16 * C_IN + 255) / 256, 256, 0, stream>>>(obj_w, delta_w, wH);
  zero_losses_kernel<<<1, 1, 0, stream>>>(out_losses);

  const int Hs[5]   = {128, 64, 32, 16, 8};
  const int Ws[5]   = {256, 128, 64, 32, 16};
  const int offs[5] = {0, 98304, 122880, 129024, 130560};
  for (int l = 0; l < 5; ++l) {
    dim3 grid((Hs[l] * Ws[l] + 15) / 16, N_IMAGES);
    rpn_head_wmma_kernel<<<grid, 256, 0, stream>>>(
        feats[l], wA, conv_b, wH, obj_b, delta_b,
        logits, deltas, Hs[l], Ws[l], offs[l]);
  }

  proposals_kernel<<<(N_IMAGES * R_TOT + 255) / 256, 256, 0, stream>>>(
      logits, deltas, boxes, scores);

  topk_nms_kernel<<<N_IMAGES, 256, 0, stream>>>(scores, boxes, out_boxes, out_scores);

  dim3 lgrid((R_TOT + 255) / 256, N_IMAGES);
  loss_kernel<<<lgrid, 256, 0, stream>>>(gt, logits, deltas, out_losses);
}
